// StructuredAttention_25898652795078
// MI455X (gfx1250) — matrix operand order repaired and checked by reading, not compiled, not run on hardware
//
#include <hip/hip_runtime.h>

typedef unsigned short u16;
typedef unsigned int   u32;

typedef __attribute__((ext_vector_type(16))) __bf16 v16bf;
typedef __attribute__((ext_vector_type(8)))  float  v8f;

union Frag { uint4 q[2]; v16bf v; };

#define BATCH 16
#define LQ 2048
#define LK 2048
#define DIM 1024
#define SCALE 0.03125f            // 1024^-0.5
#define NEG_INF_V (-1.0e9f)

__device__ __forceinline__ u16 f2bf(float f) {
  u32 x = __float_as_uint(f);
  u32 r = x + 0x7FFFu + ((x >> 16) & 1u);   // round-to-nearest-even
  return (u16)(r >> 16);
}
__device__ __forceinline__ u32 pk2(float a, float b) {
  return (u32)f2bf(a) | ((u32)f2bf(b) << 16);
}

// fast tanh: gfx1250 has V_TANH_F32 (transcendental unit)
__device__ __forceinline__ float fast_tanh(float x) {
#if __has_builtin(__builtin_amdgcn_tanhf)
  return __builtin_amdgcn_tanhf(x);
#elif __has_builtin(__builtin_amdgcn_tanh_f32)
  return __builtin_amdgcn_tanh_f32(x);
#else
  return tanhf(x);
#endif
}

// LDS byte-offset of a generic pointer into __shared__ (addrspacecast -> AS3)
__device__ __forceinline__ u32 lds_addr(const void* p) {
  return (u32)(unsigned long long)(__attribute__((address_space(3))) const void*)p;
}

// gfx1250 async global -> LDS copy, 16 bytes, tracked by ASYNCcnt
__device__ __forceinline__ void g2l_async_b128(const void* gsrc, u32 ldsoff) {
  asm volatile("global_load_async_to_lds_b128 %0, %1, off"
               :: "v"(ldsoff), "v"(gsrc) : "memory");
}
__device__ __forceinline__ void wait_async8() {
#if __has_builtin(__builtin_amdgcn_s_wait_asynccnt)
  __builtin_amdgcn_s_wait_asynccnt(8);
#else
  asm volatile("s_wait_asynccnt 0x8" ::: "memory");
#endif
}
__device__ __forceinline__ void wait_async0() {
#if __has_builtin(__builtin_amdgcn_s_wait_asynccnt)
  __builtin_amdgcn_s_wait_asynccnt(0);
#else
  asm volatile("s_wait_asynccnt 0x0" ::: "memory");
#endif
}

// ---------------- f32 -> bf16 elementwise convert (8 elems/thread) ----------
__global__ __launch_bounds__(256) void cvt_bf16_kernel(const float* __restrict__ s,
                                                       u16* __restrict__ d) {
  size_t i0 = ((size_t)blockIdx.x * 256 + threadIdx.x) * 8;
  float4 f0 = *(const float4*)(s + i0);
  float4 f1 = *(const float4*)(s + i0 + 4);
  uint4 u;
  u.x = pk2(f0.x, f0.y); u.y = pk2(f0.z, f0.w);
  u.z = pk2(f1.x, f1.y); u.w = pk2(f1.z, f1.w);
  *(uint4*)(d + i0) = u;
}

// ---------------- V [B][LK][DIM] f32  ->  Vt [B][DIM][LK] bf16 --------------
__global__ __launch_bounds__(256) void vtrans_kernel(const float* __restrict__ V,
                                                     u16* __restrict__ Vt) {
  __shared__ u16 t[64][66];
  const int b = blockIdx.z, k0 = blockIdx.y * 64, d0 = blockIdx.x * 64;
  const int tid = threadIdx.x;
  const float* src = V + ((size_t)b * LK + k0) * DIM + d0;
#pragma unroll
  for (int i = 0; i < 16; ++i) {
    int idx = tid + i * 256;
    int r = idx >> 6, c = idx & 63;                 // coalesced f32 reads
    t[c][r] = f2bf(src[(size_t)r * DIM + c]);
  }
  __syncthreads();
  u16* dst = Vt + ((size_t)b * DIM + d0) * LK + k0;
#pragma unroll
  for (int i = 0; i < 16; ++i) {
    int idx = tid + i * 256;
    int r = idx >> 6, c = idx & 63;                 // coalesced bf16 writes
    dst[(size_t)r * LK + c] = t[r][c];
  }
}

// ---------------- 128x128 tiled bf16 WMMA GEMM, async double-buffered -------
// C[M,N] (+epilogue) = A[M,K] * B[N,K]^T   (both operands K-contiguous bf16)
// EPI: 0 = tanh -> bf16 store, 1 = f32 store, 2 = f32 nontemporal store.
template <int EPI>
__global__ __launch_bounds__(256) void gemm_kernel(
    const u16* __restrict__ Ag, const u16* __restrict__ Bg, void* __restrict__ Cg,
    int N, int K, size_t sA, size_t sB, size_t sC) {
  __shared__ __align__(16) u16 lA[2][128][72];   // 64 bf16 K-slab + 8 pad, x2 buf
  __shared__ __align__(16) u16 lB[2][128][72];

  const int tid = threadIdx.x, lane = tid & 31, wave = tid >> 5;
  const int wm = wave >> 2, wn = wave & 3;          // 2x4 wave grid
  const int hs = (lane >> 4) & 1;                   // lane half-select
  const int rc = lane & 15;
  const int b = blockIdx.z;
  const int m0 = blockIdx.y * 128, n0 = blockIdx.x * 128;

  // staging geometry: per thread 4 chunks of 16B per tile; row = r0+32i, col ch
  const int r0 = tid >> 3;
  const int ch = (tid & 7) * 8;                     // bf16 element offset in slab
  const u16* aSrc = Ag + (size_t)b * sA + (size_t)(m0 + r0) * K + ch;
  const u16* bSrc = Bg + (size_t)b * sB + (size_t)(n0 + r0) * K + ch;
  u32 aOff[2], bOff[2];
  aOff[0] = lds_addr(&lA[0][r0][ch]); aOff[1] = lds_addr(&lA[1][r0][ch]);
  bOff[0] = lds_addr(&lB[0][r0][ch]); bOff[1] = lds_addr(&lB[1][r0][ch]);
  const u32 ldsRowStep = 32u * 72u * 2u;            // 32 rows in bytes

  v8f acc[4][2] = {};

  // issue first slab
#pragma unroll
  for (int i = 0; i < 4; ++i) {
    g2l_async_b128(aSrc + (size_t)i * 32 * K, aOff[0] + (u32)i * ldsRowStep);
    g2l_async_b128(bSrc + (size_t)i * 32 * K, bOff[0] + (u32)i * ldsRowStep);
  }

  // compute one 64-wide K-slab from buffer `cur`
  auto compute = [&](int cur) {
    const u16 (*cA)[72] = lA[cur];
    const u16 (*cB)[72] = lB[cur];
#pragma unroll
    for (int k0 = 0; k0 < 64; k0 += 32) {
      Frag fa[4], fb[2];
#pragma unroll
      for (int j = 0; j < 2; ++j) {                 // B: lanes16-31 take K+16..31
        int col = wn * 32 + j * 16 + rc;
        int kx = k0 + hs * 16;
        fb[j].q[0] = *(const uint4*)&cB[col][kx];
        fb[j].q[1] = *(const uint4*)&cB[col][kx + 8];
      }
#pragma unroll
      for (int i = 0; i < 4; ++i) {                 // A: lanes16-31 take K+16..31
        int row = wm * 64 + i * 16 + rc;
        int ka = k0 + hs * 16;
        fa[i].q[0] = *(const uint4*)&cA[row][ka];
        fa[i].q[1] = *(const uint4*)&cA[row][ka + 8];
      }
#pragma unroll
      for (int i = 0; i < 4; ++i)
#pragma unroll
        for (int j = 0; j < 2; ++j)
          acc[i][j] = __builtin_amdgcn_wmma_f32_16x16x32_bf16(
              false, fa[i].v, false, fb[j].v, (short)0, acc[i][j], false, false);
    }
  };

  // steady state: branch-free (prefetch next, wait current, compute)
  int cur = 0;
  for (int kb = 0; kb < K - 64; kb += 64) {
    const int nxt = cur ^ 1;
#pragma unroll
    for (int i = 0; i < 4; ++i) {
      g2l_async_b128(aSrc + (size_t)(kb + 64) + (size_t)i * 32 * K,
                     aOff[nxt] + (u32)i * ldsRowStep);
      g2l_async_b128(bSrc + (size_t)(kb + 64) + (size_t)i * 32 * K,
                     bOff[nxt] + (u32)i * ldsRowStep);
    }
    wait_async8();                                  // current slab's 8 are done
    __syncthreads();
    compute(cur);
    __syncthreads();
    cur ^= 1;
  }
  // last slab (no new issue)
  wait_async0();
  __syncthreads();
  compute(cur);

  // ---- epilogue: C layout M = r + 8*hs, N = lane%16 ----
  const int gmB = m0 + wm * 64, gnB = n0 + wn * 32;
#pragma unroll
  for (int i = 0; i < 4; ++i)
#pragma unroll
    for (int j = 0; j < 2; ++j)
#pragma unroll
      for (int r = 0; r < 8; ++r) {
        int gm = gmB + i * 16 + hs * 8 + r;
        int gn = gnB + j * 16 + rc;
        float vv = acc[i][j][r];
        if (EPI == 0) {
          u16* C = (u16*)Cg + (size_t)b * sC;
          C[(size_t)gm * N + gn] = f2bf(fast_tanh(vv));
        } else if (EPI == 1) {
          float* C = (float*)Cg + (size_t)b * sC;
          C[(size_t)gm * N + gn] = vv;
        } else {
          float* C = (float*)Cg + (size_t)b * sC;
          __builtin_nontemporal_store(vv, &C[(size_t)gm * N + gn]);
        }
      }
}

// -------- scale + mask + softmax over rows of 2048; writes f32 + bf16 -------
__global__ __launch_bounds__(256) void softmax_kernel(float* __restrict__ att,
                                                      u16* __restrict__ attb,
                                                      const int* __restrict__ kidx) {
  const int row = blockIdx.x;
  const int b = row >> 11;                     // row / LQ
  float* p = att + (size_t)row * LK;
  const int* mk = kidx + (size_t)b * LK;
  const int tid = threadIdx.x;
  const int k0 = tid * 8;

  float v[8];
  float4 a0 = *(const float4*)(p + k0);
  float4 a1 = *(const float4*)(p + k0 + 4);
  v[0]=a0.x; v[1]=a0.y; v[2]=a0.z; v[3]=a0.w;
  v[4]=a1.x; v[5]=a1.y; v[6]=a1.z; v[7]=a1.w;

  float mx = -3.4e38f;
#pragma unroll
  for (int i = 0; i < 8; ++i) {
    v[i] = (mk[k0 + i] == 0) ? NEG_INF_V : v[i] * SCALE;
    mx = fmaxf(mx, v[i]);
  }
  __shared__ float red[256];
  red[tid] = mx; __syncthreads();
  for (int s = 128; s > 0; s >>= 1) {
    if (tid < s) red[tid] = fmaxf(red[tid], red[tid + s]);
    __syncthreads();
  }
  const float m = red[0];
  __syncthreads();

  float sum = 0.f;
#pragma unroll
  for (int i = 0; i < 8; ++i) { v[i] = __expf(v[i] - m); sum += v[i]; }
  red[tid] = sum; __syncthreads();
  for (int s = 128; s > 0; s >>= 1) {
    if (tid < s) red[tid] += red[tid + s];
    __syncthreads();
  }
  const float inv = 1.0f / red[0];
#pragma unroll
  for (int i = 0; i < 8; ++i) v[i] *= inv;

  *(float4*)(p + k0)     = make_float4(v[0], v[1], v[2], v[3]);
  *(float4*)(p + k0 + 4) = make_float4(v[4], v[5], v[6], v[7]);

  uint4 u;                                    // bf16 copy for the att@V GEMM
  u.x = pk2(v[0], v[1]); u.y = pk2(v[2], v[3]);
  u.z = pk2(v[4], v[5]); u.w = pk2(v[6], v[7]);
  *(uint4*)(attb + (size_t)row * LK + k0) = u;
}

extern "C" void kernel_launch(void* const* d_in, const int* in_sizes, int n_in,
                              void* d_out, int out_size, void* d_ws, size_t ws_size,
                              hipStream_t stream) {
  const float* q   = (const float*)d_in[0];
  const float* k   = (const float*)d_in[1];
  const float* v   = (const float*)d_in[2];
  const int*   kli = (const int*)d_in[3];
  const float* W   = (const float*)d_in[4];

  float* out = (float*)d_out;                              // [B, LQ, DIM]
  float* att = (float*)d_out + (size_t)BATCH * LQ * DIM;   // [B, LQ, LK]

  // ws layout (bytes): [qb 67M][kb 67M][vtb 67M][wb 2M][ktb 67M] ≈ 258 MB.
  // attb (bf16 att, 134M) overlays qb+kb after they are dead (post-gemm2).
  char* ws = (char*)d_ws;
  const size_t szQ = (size_t)BATCH * LQ * DIM;  // 33,554,432 elements
  const size_t szW = (size_t)DIM * DIM;
  u16* qb   = (u16*)(ws);
  u16* kb   = (u16*)(ws + szQ * 2);
  u16* vtb  = (u16*)(ws + szQ * 4);
  u16* wb   = (u16*)(ws + szQ * 6);
  u16* ktb  = (u16*)(ws + szQ * 6 + szW * 2);
  u16* attb = (u16*)(ws);                       // reuses qb+kb space

  cvt_bf16_kernel<<<(unsigned)(szQ / 2048), 256, 0, stream>>>(q, qb);
  cvt_bf16_kernel<<<(unsigned)(szQ / 2048), 256, 0, stream>>>(k, kb);
  cvt_bf16_kernel<<<(unsigned)(szW / 2048), 256, 0, stream>>>(W, wb);
  vtrans_kernel<<<dim3(DIM / 64, LK / 64, BATCH), 256, 0, stream>>>(v, vtb);

  // kt = tanh(K @ W^T): M=LK, N=DIM, Kd=DIM  (W shared across batch: sB=0)
  gemm_kernel<0><<<dim3(DIM / 128, LK / 128, BATCH), 256, 0, stream>>>(
      kb, wb, ktb, DIM, DIM, (size_t)LK * DIM, (size_t)0, (size_t)LK * DIM);

  // score = Q @ kt^T: M=LQ, N=LK, Kd=DIM  -> raw f32 into att region
  gemm_kernel<1><<<dim3(LK / 128, LQ / 128, BATCH), 256, 0, stream>>>(
      qb, ktb, att, LK, DIM, (size_t)LQ * DIM, (size_t)LK * DIM, (size_t)LQ * LK);

  // scale + mask + softmax, in place on att; also emit bf16 att
  softmax_kernel<<<BATCH * LQ, 256, 0, stream>>>(att, attb, kli);

  // out = att @ V: M=LQ, N=DIM, Kd=LK  (A = attb bf16; B = Vt, NT stores)
  gemm_kernel<2><<<dim3(DIM / 128, LQ / 128, BATCH), 256, 0, stream>>>(
      attb, vtb, out, DIM, LK, (size_t)LQ * LK, (size_t)DIM * LK, (size_t)LQ * DIM);
}
